// GNNPotentials_77326591197639
// MI455X (gfx1250) — matrix operand order repaired and checked by reading, not compiled, not run on hardware
//
#include <hip/hip_runtime.h>

typedef __attribute__((ext_vector_type(16))) _Float16 v16h;
typedef __attribute__((ext_vector_type(8)))  float    v8f;

#define N_ATOMS   2048
#define NRBF      16
#define HID       64
#define CUTOFF2   25.0f
#define GAMMA_RBF 10.0f
#define JB_COUNT  (N_ATOMS / 16)          // 128 j-chunks per row
#define TILES     (N_ATOMS * JB_COUNT)    // 262144 (i, jBlock) tiles
#define BLOCKS_A  1024
#define THREADS_A 256
#define NWAVES    (BLOCKS_A * THREADS_A / 32)  // 8192 waves, 32 tiles each

// Fast SiLU: x * rcp(1 + exp(-x)).  v_exp_f32 + v_rcp_f32 are TRANS ops that
// co-execute with the matrix pipe; avoids the IEEE v_div_* expansion.
__device__ __forceinline__ float silu_fast(float x) {
    return x * __builtin_amdgcn_rcpf(1.0f + __expf(-x));
}

__device__ __forceinline__ unsigned ballot32(bool p) {
#if defined(__has_builtin)
#if __has_builtin(__builtin_amdgcn_ballot_w32)
    return (unsigned)__builtin_amdgcn_ballot_w32(p);
#else
    return (unsigned)__ballot(p);
#endif
#else
    return (unsigned)__ballot(p);
#endif
}

// One wave handles 16 (i,j) pairs at a time: A-tile = 16 edges x 32 K (RBF,
// zero-padded), B-tiles = W1 zero-padded to 32x16 per 16-wide HID slice.
// Wave-uniform tile skipping keeps EXEC all-ones across the WMMA ops.
__global__ void __launch_bounds__(THREADS_A)
gnn_edge_energy_kernel(const float* __restrict__ q,
                       const float* __restrict__ cell,
                       const float* __restrict__ W1,
                       const float* __restrict__ b1,
                       const float* __restrict__ W2,
                       float* __restrict__ partials,
                       unsigned* __restrict__ counts)
{
    const int  lane    = threadIdx.x & 31;
    const int  wave    = (int)((blockIdx.x * blockDim.x + threadIdx.x) >> 5);
    const int  hid_lo  = lane & 15;
    const bool hiHalf  = lane >= 16;

    const float cx = cell[0], cy = cell[1], cz = cell[2];
    const float hx = 0.5f * cx, hy = 0.5f * cy, hz = 0.5f * cz;

    // ---- per-wave constants: B tiles (f16, K padded 16->32), b1, W2 slices ----
    v16h  Bt[4];
    float b1v[4], w2v[4];
#pragma unroll
    for (int nt = 0; nt < 4; ++nt) {
        const int hid = nt * 16 + hid_lo;
#pragma unroll
        for (int e = 0; e < 16; ++e) {
            // half h, element e<8 <-> K=8h+e; elements 8..15 hold K>=16 -> zero pad
            Bt[nt][e] = (e >= 8) ? (_Float16)0.0f : (_Float16)W1[((hiHalf ? 8 : 0) + e) * HID + hid];
        }
        b1v[nt] = b1[hid];
        w2v[nt] = W2[hid];
    }

    const float kbase = hiHalf ? 8.0f : 0.0f;   // this half-wave's RBF K offset
    const int   mbase = hiHalf ? 8 : 0;         // this half-wave's edge-row offset in D

    float    s_acc = 0.0f;
    unsigned e_cnt = 0;                          // wave-uniform valid-pair count

    for (int tile = wave; tile < TILES; tile += NWAVES) {
        const int i  = tile >> 7;                // row atom
        const int jb = tile & (JB_COUNT - 1);    // 16-wide column chunk
        if (jb * 16 + 15 <= i) continue;         // chunk entirely in lower triangle (uniform)

        const int j = jb * 16 + (lane & 15);

        // minimum-image displacement (matches reference offsets formula)
        const float qix = q[i * 3 + 0], qiy = q[i * 3 + 1], qiz = q[i * 3 + 2];
        float dx = q[j * 3 + 0] - qix;
        float dy = q[j * 3 + 1] - qiy;
        float dz = q[j * 3 + 2] - qiz;
        dx += (dx >= hx) ? -cx : ((dx < -hx) ? cx : 0.0f);
        dy += (dy >= hy) ? -cy : ((dy < -hy) ? cy : 0.0f);
        dz += (dz >= hz) ? -cz : ((dz < -hz) ? cz : 0.0f);
        const float dsq = dx * dx + dy * dy + dz * dz;

        const bool valid = (j > i) && (dsq < CUTOFF2) && (dsq != 0.0f);
        const unsigned mask = ballot32(valid) & 0xFFFFu;  // per-pair mask (lanes 16-31 mirror)
        if (mask == 0u) continue;                          // uniform skip
        e_cnt += (unsigned)__popc((int)mask);

        const float d = valid ? sqrtf(dsq) : 1.0e3f;       // far dummy -> rbf == 0

        // ---- A tile: 16x32 f16, lane holds row m=lane%16, K = kbase+0..7 ----
        v16h A;
#pragma unroll
        for (int e = 0; e < 8; ++e) {
            const float mu = (kbase + (float)e) * (5.0f / 15.0f);
            const float t  = d - mu;
            A[e] = (_Float16)__expf(-GAMMA_RBF * t * t);
        }
#pragma unroll
        for (int e = 8; e < 16; ++e) A[e] = (_Float16)0.0f;  // K = 16..31 zero pad

        // ---- 4 WMMAs cover HID=64; fuse bias + SiLU + W2 dot, branchless gate ----
#pragma unroll
        for (int nt = 0; nt < 4; ++nt) {
            v8f c = {};
            c = __builtin_amdgcn_wmma_f32_16x16x32_f16(
                    /*neg_a=*/false, A, /*neg_b=*/false, Bt[nt],
                    /*c_mod=*/(short)0, c, /*reuse_a=*/false, /*reuse_b=*/false);
#pragma unroll
            for (int r = 0; r < 8; ++r) {
                const int m = mbase + r;                     // edge row of this D VGPR
                // branchless per-row gate: v_cndmask instead of exec-mask branch
                const float gate_w2 = ((mask >> m) & 1u) ? w2v[nt] : 0.0f;
                s_acc += silu_fast(c[r] + b1v[nt]) * gate_w2;
            }
        }
    }

    // deterministic intra-wave reduction (fixed shuffle pattern, wave32)
#pragma unroll
    for (int off = 16; off >= 1; off >>= 1)
        s_acc += __shfl_xor(s_acc, off, 32);

    if (lane == 0) {
        partials[wave] = s_acc;   // every wave writes its fixed slot
        counts[wave]   = e_cnt;
    }
}

__global__ void gnn_reduce_kernel(const float* __restrict__ partials,
                                  const unsigned* __restrict__ counts,
                                  const float* __restrict__ b2,
                                  float* __restrict__ out)
{
    __shared__ float    sf[256];
    __shared__ unsigned su[256];
    float    s = 0.0f;
    unsigned c = 0u;
    for (int idx = (int)threadIdx.x; idx < NWAVES; idx += 256) {
        s += partials[idx];
        c += counts[idx];
    }
    sf[threadIdx.x] = s;
    su[threadIdx.x] = c;
    __syncthreads();
    for (int off = 128; off > 0; off >>= 1) {
        if ((int)threadIdx.x < off) {
            sf[threadIdx.x] += sf[threadIdx.x + off];
            su[threadIdx.x] += su[threadIdx.x + off];
        }
        __syncthreads();
    }
    if (threadIdx.x == 0)
        out[0] = sf[0] + b2[0] * (float)su[0];
}

extern "C" void kernel_launch(void* const* d_in, const int* in_sizes, int n_in,
                              void* d_out, int out_size, void* d_ws, size_t ws_size,
                              hipStream_t stream) {
    (void)in_sizes; (void)n_in; (void)out_size; (void)ws_size;
    const float* q    = (const float*)d_in[0];   // [2048,3]
    const float* cell = (const float*)d_in[1];   // [3]
    const float* W1   = (const float*)d_in[2];   // [16,64]
    const float* b1   = (const float*)d_in[3];   // [64]
    const float* W2   = (const float*)d_in[4];   // [64,1]
    const float* b2   = (const float*)d_in[5];   // [1]

    float*    partials = (float*)d_ws;                                      // 8192 floats
    unsigned* counts   = (unsigned*)((char*)d_ws + NWAVES * sizeof(float)); // 8192 uints

    gnn_edge_energy_kernel<<<BLOCKS_A, THREADS_A, 0, stream>>>(
        q, cell, W1, b1, W2, partials, counts);
    gnn_reduce_kernel<<<1, 256, 0, stream>>>(partials, counts, b2, (float*)d_out);
}
